// BertContrastivePredictor_36558761624315
// MI455X (gfx1250) — matrix operand order repaired and checked by reading, not compiled, not run on hardware
//
#include <hip/hip_runtime.h>
#include <hip/hip_bf16.h>
#include <math.h>

// ---------------- constants from the reference ----------------
#define Bb 32
#define Ss 512
#define Dd 1024
#define Hh 512
#define Kk 16
#define Ll 16
#define NSs 16
#define NTt 16
#define OUTC 2080           // 1024 slot_feats + 1024 context + 32 labels
#define SMOOTHC 0.1f
#define EPSC 1e-8f

typedef unsigned short ub16;
typedef __bf16 bf16_t;
typedef bf16_t v16bf __attribute__((ext_vector_type(16)));
typedef float  v8f   __attribute__((ext_vector_type(8)));
typedef ub16   ub16x8 __attribute__((ext_vector_type(8)));

union Frag { v16bf v; ub16x8 h[2]; ub16 s[16]; };

__device__ __forceinline__ ub16 f2bf(float f) {
    union { float f; unsigned u; } c; c.f = f;
    unsigned r = c.u + 0x7FFFu + ((c.u >> 16) & 1u);   // round-to-nearest-even
    return (ub16)(r >> 16);
}
__device__ __forceinline__ float sigm(float x) { return 1.f / (1.f + expf(-x)); }

// ---------------------------------------------------------------------------
// Register-blocked bf16 WMMA GEMM for the big matmuls.
// Each wave computes a 32(M) x 64(N) tile: 2 A-frags, 4 B-frags, 8 WMMAs/k-step.
// Block = 4 waves covering 32(M) x 256(N).  A row-major [M,K]; B in weight
// layout BT[N,K] row-major.  Requires M%32==0, N%256==0, K%32==0.
// grid = (N/256, M/32).
// ---------------------------------------------------------------------------
__global__ __launch_bounds__(128) void gemm_bf16_wmma_blk(
    const ub16* __restrict__ A, int lda,
    const ub16* __restrict__ Bm, int ldb,
    float* __restrict__ C, int ldc,
    int Kd, const float* __restrict__ bias, int accum)
{
    const int lane = threadIdx.x & 31;
    const int wid  = threadIdx.x >> 5;
    const int hf   = lane >> 4;        // lane half
    const int ml   = lane & 15;
    const int nbase = blockIdx.x * 256 + wid * 64;
    const int mbase = blockIdx.y * 32;

    const ub16* arow0 = A + (size_t)(mbase + ml) * lda + hf * 8;
    const ub16* arow1 = arow0 + (size_t)16 * lda;
    const ub16* brow[4];
#pragma unroll
    for (int j = 0; j < 4; ++j)
        brow[j] = Bm + (size_t)(nbase + j * 16 + ml) * ldb + hf * 8;

    v8f acc[2][4];
    if (accum) {
#pragma unroll
        for (int i = 0; i < 2; ++i)
#pragma unroll
            for (int j = 0; j < 4; ++j)
#pragma unroll
                for (int r = 0; r < 8; ++r)
                    acc[i][j][r] = C[(size_t)(mbase + i * 16 + hf * 8 + r) * ldc
                                     + nbase + j * 16 + ml];
    } else if (bias) {
#pragma unroll
        for (int j = 0; j < 4; ++j) {
            float bv = bias[nbase + j * 16 + ml];
#pragma unroll
            for (int i = 0; i < 2; ++i)
#pragma unroll
                for (int r = 0; r < 8; ++r) acc[i][j][r] = bv;
        }
    } else {
#pragma unroll
        for (int i = 0; i < 2; ++i)
#pragma unroll
            for (int j = 0; j < 4; ++j)
#pragma unroll
                for (int r = 0; r < 8; ++r) acc[i][j][r] = 0.f;
    }

    for (int k = 0; k < Kd; k += 32) {
        Frag fa0, fa1, fb[4];
        fa0.h[0] = *(const ub16x8*)(arow0 + k);
        fa0.h[1] = *(const ub16x8*)(arow0 + k + 16);
        fa1.h[0] = *(const ub16x8*)(arow1 + k);
        fa1.h[1] = *(const ub16x8*)(arow1 + k + 16);
#pragma unroll
        for (int j = 0; j < 4; ++j) {
            fb[j].h[0] = *(const ub16x8*)(brow[j] + k);
            fb[j].h[1] = *(const ub16x8*)(brow[j] + k + 16);
        }
        __builtin_prefetch(arow0 + k + 32, 0, 1);
        __builtin_prefetch(arow1 + k + 32, 0, 1);
#pragma unroll
        for (int j = 0; j < 4; ++j) {
            acc[0][j] = __builtin_amdgcn_wmma_f32_16x16x32_bf16(
                false, fa0.v, false, fb[j].v, (short)0, acc[0][j], false, false);
            acc[1][j] = __builtin_amdgcn_wmma_f32_16x16x32_bf16(
                false, fa1.v, false, fb[j].v, (short)0, acc[1][j], false, false);
        }
    }
#pragma unroll
    for (int i = 0; i < 2; ++i)
#pragma unroll
        for (int j = 0; j < 4; ++j)
#pragma unroll
            for (int r = 0; r < 8; ++r)
                C[(size_t)(mbase + i * 16 + hf * 8 + r) * ldc + nbase + j * 16 + ml]
                    = acc[i][j][r];
}

// ---------------------------------------------------------------------------
// Generic wave-level bf16 WMMA GEMM (one 16x16 tile per wave) for odd shapes.
// bRowMajor=1: B is [K,N] row-major (strided loads; tiny GEMMs only).
// grid = (N/16, M/16, batch), block = 32.
// ---------------------------------------------------------------------------
__global__ __launch_bounds__(32) void gemm_bf16_wmma(
    const ub16* __restrict__ A, long long aBatch, int lda,
    const ub16* __restrict__ Bm, long long bBatch, int ldb,
    float* __restrict__ C, long long cBatch, int ldc,
    int Kd, const float* __restrict__ bias, int accum, int bRowMajor)
{
    const int lane = threadIdx.x;
    const int nt = blockIdx.x, mt = blockIdx.y, bz = blockIdx.z;
    A  += (long long)bz * aBatch;
    Bm += (long long)bz * bBatch;
    C  += (long long)bz * cBatch;

    const int hf = lane >> 4;
    const int kb = hf * 16;
    const ub16* arow = A + (size_t)(mt * 16 + (lane & 15)) * lda + hf * 16;

    const int cn  = nt * 16 + (lane & 15);
    const int cm0 = mt * 16 + hf * 8;

    v8f acc;
    if (accum) {
#pragma unroll
        for (int r = 0; r < 8; ++r) acc[r] = C[(size_t)(cm0 + r) * ldc + cn];
    } else if (bias) {
        float bv = bias[cn];
#pragma unroll
        for (int r = 0; r < 8; ++r) acc[r] = bv;
    } else {
#pragma unroll
        for (int r = 0; r < 8; ++r) acc[r] = 0.f;
    }

    if (!bRowMajor) {
        const ub16* brow = Bm + (size_t)(nt * 16 + (lane & 15)) * ldb + kb;
        for (int k = 0; k < Kd; k += 32) {
            Frag fa, fb;
            fa.h[0] = *(const ub16x8*)(arow + k);
            fa.h[1] = *(const ub16x8*)(arow + k + 8);
            fb.h[0] = *(const ub16x8*)(brow + k);
            fb.h[1] = *(const ub16x8*)(brow + k + 8);
            acc = __builtin_amdgcn_wmma_f32_16x16x32_bf16(
                false, fa.v, false, fb.v, (short)0, acc, false, false);
        }
    } else {
        for (int k = 0; k < Kd; k += 32) {
            Frag fa, fb;
            fa.h[0] = *(const ub16x8*)(arow + k);
            fa.h[1] = *(const ub16x8*)(arow + k + 8);
#pragma unroll
            for (int j = 0; j < 16; ++j)
                fb.s[j] = Bm[(size_t)(k + kb + j) * ldb + nt * 16 + (lane & 15)];
            acc = __builtin_amdgcn_wmma_f32_16x16x32_bf16(
                false, fa.v, false, fb.v, (short)0, acc, false, false);
        }
    }
#pragma unroll
    for (int r = 0; r < 8; ++r) C[(size_t)(cm0 + r) * ldc + cn] = acc[r];
}

// ------------------------------ pointwise -----------------------------------
__global__ void k_f32_to_bf16(const float* __restrict__ s, ub16* __restrict__ d, int n) {
    int i = blockIdx.x * blockDim.x + threadIdx.x;
    if (i < n) d[i] = f2bf(s[i]);
}
__global__ void k_zero(float* p, int n) {
    int i = blockIdx.x * blockDim.x + threadIdx.x;
    if (i < n) p[i] = 0.f;
}
__global__ void k_zero_bf16(ub16* p, int n) {
    int i = blockIdx.x * blockDim.x + threadIdx.x;
    if (i < n) p[i] = 0;
}
__global__ void k_bias_sum(const float* a, const float* b, float* o, int n) {
    int i = blockIdx.x * blockDim.x + threadIdx.x;
    if (i < n) o[i] = a[i] + b[i];
}
// spansBF[(m*L+l)*D+d] = hidden[b, span_idx[b,k,l], d]  (bf16)
__global__ void k_gather_spans(const float* __restrict__ hidden,
                               const int* __restrict__ span_idx,
                               ub16* __restrict__ spansBF) {
    int i = blockIdx.x * blockDim.x + threadIdx.x;           // < B*K*L*D
    if (i >= Bb * Kk * Ll * Dd) return;
    int d = i & (Dd - 1);
    int r = i >> 10;                 // m*L + l
    int m = r >> 4;                  // span row (b*K+k)
    int b = m >> 4;
    int sidx = span_idx[r];
    spansBF[i] = f2bf(hidden[((size_t)b * Ss + sidx) * Dd + d]);
}
// LSTM gates for one timestep (512 spans x H)
__global__ void k_lstm_point(const float* __restrict__ G,
                             float* __restrict__ Cst, float* __restrict__ Hst,
                             ub16* __restrict__ Hbf, float* __restrict__ Hsum,
                             const float* __restrict__ mask,
                             int xidx, int dirOff, int isFirst) {
    int i = blockIdx.x * blockDim.x + threadIdx.x;           // < 512*512
    if (i >= Bb * Kk * Hh) return;
    int m = i >> 9, h = i & (Hh - 1);
    const float* g = G + (size_t)m * (4 * Hh);
    float gi = g[h], gf = g[Hh + h], gg = g[2 * Hh + h], go = g[3 * Hh + h];
    float cprev = isFirst ? 0.f : Cst[i];
    float hprev = isFirst ? 0.f : Hst[i];
    float mt = mask[m * Ll + xidx];
    float cn = sigm(gf) * cprev + sigm(gi) * tanhf(gg);
    float hn = sigm(go) * tanhf(cn);
    float hN = mt * hn + (1.f - mt) * hprev;
    float cN = mt * cn + (1.f - mt) * cprev;
    Cst[i] = cN; Hst[i] = hN; Hbf[i] = f2bf(hN);
    Hsum[(size_t)m * (2 * Hh) + dirOff + h] += hN * mt;      // ys = h * m, summed over L
}
__global__ void k_copy_slotfeats(const float* __restrict__ Hsum, float* __restrict__ out) {
    int i = blockIdx.x * blockDim.x + threadIdx.x;           // < 512*1024
    if (i >= Bb * Kk * 2 * Hh) return;
    int row = i >> 10, h = i & 1023;
    out[(size_t)row * OUTC + h] = Hsum[i];
}
// attnBF[b,k,s] = cmask ? attnT[b,s,k] : 0   (bf16)
__global__ void k_attn_mask(const float* __restrict__ attnT,
                            const int* __restrict__ start, const int* __restrict__ endp,
                            const int* __restrict__ length, ub16* __restrict__ attnBF) {
    int i = blockIdx.x * blockDim.x + threadIdx.x;           // < B*K*S
    if (i >= Bb * Kk * Ss) return;
    int s = i & (Ss - 1);
    int k = (i >> 9) & (Kk - 1);
    int b = i >> 13;
    int row = b * Kk + k;
    bool keep = (s < start[row]) || ((s > endp[row]) && (s < length[b]));
    float v = keep ? attnT[((size_t)b * Ss + s) * Kk + k] : 0.f;
    attnBF[i] = f2bf(v);
}
__global__ void k_tanh_bf16(const float* __restrict__ in, ub16* __restrict__ out, int n) {
    int i = blockIdx.x * blockDim.x + threadIdx.x;
    if (i < n) out[i] = f2bf(tanhf(in[i]));
}
// one wave per row: L2 norm of 2048-wide row
__global__ __launch_bounds__(32) void k_rownorm(const float* __restrict__ scat,
                                                float* __restrict__ norms) {
    int row = blockIdx.x, lane = threadIdx.x;
    float s = 0.f;
    for (int j = lane; j < 2 * Dd; j += 32) {
        float v = scat[(size_t)row * (2 * Dd) + j];
        s += v * v;
    }
#pragma unroll
    for (int o = 16; o > 0; o >>= 1) s += __shfl_xor(s, o, 32);
    if (lane == 0) norms[row] = sqrtf(s);
}
__global__ void k_labels(const float* __restrict__ simraw, const float* __restrict__ norms,
                         const int* __restrict__ sids, float* __restrict__ out) {
    int row = blockIdx.x * blockDim.x + threadIdx.x;         // < 512
    if (row >= Bb * Kk) return;
    float sn = fmaxf(norms[row], EPSC);
    float sv[NTt]; float tot = 0.f;
#pragma unroll
    for (int t = 0; t < NTt; ++t) {
        float tn = fmaxf(norms[Bb * Kk + t], EPSC);
        sv[t] = simraw[row * NTt + t] / (sn * tn);
        tot += sv[t];
    }
    float scale = (1.f - SMOOTHC) / tot;
    int sid = sids[row];
    float* o = out + (size_t)row * OUTC + 2048;
#pragma unroll
    for (int j = 0; j < NSs; ++j) o[j] = (j == sid) ? SMOOTHC : 0.f;
#pragma unroll
    for (int t = 0; t < NTt; ++t) o[NSs + t] = sv[t] * scale;
}

// ---------------------------------------------------------------------------
extern "C" void kernel_launch(void* const* d_in, const int* in_sizes, int n_in,
                              void* d_out, int out_size, void* d_ws, size_t ws_size,
                              hipStream_t stream) {
    const float* hidden   = (const float*)d_in[0];
    const int*   span_idx = (const int*)d_in[1];
    const float* span_mask= (const float*)d_in[2];
    const int*   sstart   = (const int*)d_in[3];
    const int*   send     = (const int*)d_in[4];
    const int*   length   = (const int*)d_in[5];
    const float* slot_emb = (const float*)d_in[6];
    const float* tgt      = (const float*)d_in[7];
    const int*   sids     = (const int*)d_in[8];
    const float* Wih_f = (const float*)d_in[9],  *Whh_f = (const float*)d_in[10];
    const float* bih_f = (const float*)d_in[11], *bhh_f = (const float*)d_in[12];
    const float* Wih_b = (const float*)d_in[13], *Whh_b = (const float*)d_in[14];
    const float* bih_b = (const float*)d_in[15], *bhh_b = (const float*)d_in[16];
    const float* Wps1 = (const float*)d_in[17], *bps1 = (const float*)d_in[18];
    const float* Wps2 = (const float*)d_in[19], *bps2 = (const float*)d_in[20];
    const float* Wpc1 = (const float*)d_in[21], *bpc1 = (const float*)d_in[22];
    const float* Wpc2 = (const float*)d_in[23], *bpc2 = (const float*)d_in[24];
    float* out = (float*)d_out;

    char* w = (char*)d_ws; size_t off = 0;
    auto alloc = [&](size_t bytes) -> void* {
        void* p = w + off; off = (off + bytes + 255) & ~(size_t)255; return p;
    };
    const int MR  = Bb * Kk;          // 512 span rows
    const int MU  = MR + NTt;         // 528 valid MLP rows
    const int MUP = 544;              // padded to multiple of 32
    ub16* hiddenBF = (ub16*)alloc((size_t)Bb * Ss * Dd * 2);
    ub16* spansBF  = (ub16*)alloc((size_t)MR * Ll * Dd * 2);
    ub16* WihF = (ub16*)alloc((size_t)4 * Hh * Dd * 2);
    ub16* WhhF = (ub16*)alloc((size_t)4 * Hh * Hh * 2);
    ub16* WihB = (ub16*)alloc((size_t)4 * Hh * Dd * 2);
    ub16* WhhB = (ub16*)alloc((size_t)4 * Hh * Hh * 2);
    ub16* Wp1s = (ub16*)alloc((size_t)Dd * Dd * 2);
    ub16* Wp2s = (ub16*)alloc((size_t)Dd * Dd * 2);
    ub16* Wp1c = (ub16*)alloc((size_t)Dd * Dd * 2);
    ub16* Wp2c = (ub16*)alloc((size_t)Dd * Dd * 2);
    ub16* UBF  = (ub16*)alloc((size_t)MUP * Dd * 2);
    float* bsumF = (float*)alloc(4 * Hh * 4);
    float* bsumB = (float*)alloc(4 * Hh * 4);
    float* G    = (float*)alloc((size_t)MR * 4 * Hh * 4);
    float* Cst  = (float*)alloc((size_t)MR * Hh * 4);
    float* Hst  = (float*)alloc((size_t)MR * Hh * 4);
    ub16*  Hbf  = (ub16*)alloc((size_t)MR * Hh * 2);
    float* Hsum = (float*)alloc((size_t)MR * 2 * Hh * 4);
    ub16*  sfBF = (ub16*)alloc((size_t)MR * 2 * Hh * 2);
    float* attnT = (float*)alloc((size_t)Bb * Ss * Kk * 4);
    ub16*  attnBF= (ub16*)alloc((size_t)Bb * Kk * Ss * 2);
    float* tmp1  = (float*)alloc((size_t)MUP * Dd * 4);
    ub16*  t1BF  = (ub16*)alloc((size_t)MUP * Dd * 2);
    float* scat  = (float*)alloc((size_t)MUP * 2 * Dd * 4);
    ub16*  scatBF= (ub16*)alloc((size_t)MUP * 2 * Dd * 2);
    float* simraw= (float*)alloc((size_t)MR * NTt * 4);
    float* norms = (float*)alloc((size_t)MUP * 4);

    const int T = 256;
    auto g1 = [&](int n) { return dim3((n + T - 1) / T); };

    // ---- precision conversion / gathers ----
    k_f32_to_bf16<<<g1(Bb*Ss*Dd), T, 0, stream>>>(hidden, hiddenBF, Bb*Ss*Dd);
    k_gather_spans<<<g1(MR*Ll*Dd), T, 0, stream>>>(hidden, span_idx, spansBF);
    k_f32_to_bf16<<<g1(4*Hh*Dd), T, 0, stream>>>(Wih_f, WihF, 4*Hh*Dd);
    k_f32_to_bf16<<<g1(4*Hh*Hh), T, 0, stream>>>(Whh_f, WhhF, 4*Hh*Hh);
    k_f32_to_bf16<<<g1(4*Hh*Dd), T, 0, stream>>>(Wih_b, WihB, 4*Hh*Dd);
    k_f32_to_bf16<<<g1(4*Hh*Hh), T, 0, stream>>>(Whh_b, WhhB, 4*Hh*Hh);
    k_f32_to_bf16<<<g1(Dd*Dd), T, 0, stream>>>(Wps1, Wp1s, Dd*Dd);
    k_f32_to_bf16<<<g1(Dd*Dd), T, 0, stream>>>(Wps2, Wp2s, Dd*Dd);
    k_f32_to_bf16<<<g1(Dd*Dd), T, 0, stream>>>(Wpc1, Wp1c, Dd*Dd);
    k_f32_to_bf16<<<g1(Dd*Dd), T, 0, stream>>>(Wpc2, Wp2c, Dd*Dd);
    k_f32_to_bf16<<<g1(MR*Dd), T, 0, stream>>>(slot_emb, UBF, MR*Dd);
    k_f32_to_bf16<<<g1(NTt*Dd), T, 0, stream>>>(tgt, UBF + (size_t)MR*Dd, NTt*Dd);
    k_zero_bf16<<<g1((MUP-MU)*Dd), T, 0, stream>>>(UBF + (size_t)MU*Dd, (MUP-MU)*Dd);
    k_bias_sum<<<g1(4*Hh), T, 0, stream>>>(bih_f, bhh_f, bsumF, 4*Hh);
    k_bias_sum<<<g1(4*Hh), T, 0, stream>>>(bih_b, bhh_b, bsumB, 4*Hh);
    k_zero<<<g1(MR*2*Hh), T, 0, stream>>>(Hsum, MR*2*Hh);

    // ---- BiLSTM: 16 sequential steps per direction, blocked WMMA GEMMs ----
    for (int dir = 0; dir < 2; ++dir) {
        const ub16* Wih = dir ? WihB : WihF;
        const ub16* Whh = dir ? WhhB : WhhF;
        const float* bs = dir ? bsumB : bsumF;
        int dirOff = dir ? Hh : 0;
        for (int t = 0; t < Ll; ++t) {
            int xidx = dir ? (Ll - 1 - t) : t;
            // G = x_t @ Wih^T + (bih+bhh) :  M=512, K=1024, N=2048
            gemm_bf16_wmma_blk<<<dim3(4*Hh/256, MR/32), 128, 0, stream>>>(
                spansBF + (size_t)xidx * Dd, Ll * Dd, Wih, Dd, G, 4 * Hh, Dd, bs, 0);
            if (t > 0)   // G += h_{t-1} @ Whh^T : M=512, K=512, N=2048
                gemm_bf16_wmma_blk<<<dim3(4*Hh/256, MR/32), 128, 0, stream>>>(
                    Hbf, Hh, Whh, Hh, G, 4 * Hh, Hh, nullptr, 1);
            k_lstm_point<<<g1(MR*Hh), T, 0, stream>>>(
                G, Cst, Hst, Hbf, Hsum, span_mask, xidx, dirOff, t == 0);
        }
    }

    // ---- slot_feats -> out[:, :1024] and bf16 for attention ----
    k_copy_slotfeats<<<g1(MR*2*Hh), T, 0, stream>>>(Hsum, out);
    k_f32_to_bf16<<<g1(MR*2*Hh), T, 0, stream>>>(Hsum, sfBF, MR*2*Hh);

    // ---- attention: attnT[b] (S x K) = hidden[b] @ sf[b]^T ----
    gemm_bf16_wmma<<<dim3(Kk/16, Ss/16, Bb), 32, 0, stream>>>(
        hiddenBF, (long long)Ss * Dd, Dd,
        sfBF, (long long)Kk * Dd, Dd,
        attnT, (long long)Ss * Kk, Kk, Dd, nullptr, 0, 0);
    k_attn_mask<<<g1(Bb*Kk*Ss), T, 0, stream>>>(attnT, sstart, send, length, attnBF);
    // context[b] (K x D) = attn[b] @ hidden[b]  -> out[:, 1024:2048]
    gemm_bf16_wmma<<<dim3(Dd/16, Kk/16, Bb), 32, 0, stream>>>(
        attnBF, (long long)Kk * Ss, Ss,
        hiddenBF, (long long)Ss * Dd, Dd,
        out + 1024, (long long)Kk * OUTC, OUTC, Ss, nullptr, 0, 1);

    // ---- MLP branches on U = [slot_emb; tgt] padded to 544 x 1024 ----
    gemm_bf16_wmma_blk<<<dim3(Dd/256, MUP/32), 128, 0, stream>>>(
        UBF, Dd, Wp1s, Dd, tmp1, Dd, Dd, bps1, 0);
    k_tanh_bf16<<<g1(MUP*Dd), T, 0, stream>>>(tmp1, t1BF, MUP*Dd);
    gemm_bf16_wmma_blk<<<dim3(Dd/256, MUP/32), 128, 0, stream>>>(
        t1BF, Dd, Wp2s, Dd, scat, 2 * Dd, Dd, bps2, 0);
    gemm_bf16_wmma_blk<<<dim3(Dd/256, MUP/32), 128, 0, stream>>>(
        UBF, Dd, Wp1c, Dd, tmp1, Dd, Dd, bpc1, 0);
    k_tanh_bf16<<<g1(MUP*Dd), T, 0, stream>>>(tmp1, t1BF, MUP*Dd);
    gemm_bf16_wmma_blk<<<dim3(Dd/256, MUP/32), 128, 0, stream>>>(
        t1BF, Dd, Wp2c, Dd, scat + Dd, 2 * Dd, Dd, bpc2, 0);

    // ---- cosine sim + label smoothing -> out[:, 2048:2080] ----
    k_f32_to_bf16<<<g1(MUP*2*Dd), T, 0, stream>>>(scat, scatBF, MUP*2*Dd);
    k_rownorm<<<dim3(MU), 32, 0, stream>>>(scat, norms);
    gemm_bf16_wmma<<<dim3(NTt/16, MR/16, 1), 32, 0, stream>>>(
        scatBF, 0, 2 * Dd,
        scatBF + (size_t)MR * 2 * Dd, 0, 2 * Dd,
        simraw, 0, NTt, 2 * Dd, nullptr, 0, 0);
    k_labels<<<g1(MR), T, 0, stream>>>(simraw, norms, sids, out);
}
